// MambaSSMLayer_51668456570916
// MI455X (gfx1250) — matrix operand order repaired and checked by reading, not compiled, not run on hardware
//
#include <hip/hip_runtime.h>

#define D_MODEL    768
#define D_STATE    16
#define D_CONVK    4
#define D_INNER    1536
#define DT_RANK    48
#define DT_RANKP   64
#define BATCH      4
#define SEQ        2048
#define NTOK       (BATCH * SEQ)        // 8192
#define N_XZ       (2 * D_INNER)        // 3072
#define N_XDBL     (DT_RANK + 2 * D_STATE) // 80
#define TPB        256

typedef __bf16 bf16;
typedef __attribute__((ext_vector_type(16))) __bf16 v16bf;
typedef __attribute__((ext_vector_type(8)))  __bf16 v8bf;
typedef __attribute__((ext_vector_type(8)))  float  v8f;
typedef int v4i __attribute__((vector_size(16)));

// ---- CDNA5 async-copy-to-LDS path (guarded; falls back to ds_store pipeline)
#if defined(__has_builtin)
#if __has_builtin(__builtin_amdgcn_global_load_async_to_lds_b128)
#define HAVE_ASYNC_LDS 1
#endif
#endif

#ifdef HAVE_ASYNC_LDS
__device__ __forceinline__ void async_cp_b128(const void* g, void* l) {
  __builtin_amdgcn_global_load_async_to_lds_b128(
      (__attribute__((address_space(1))) v4i*)(g),
      (__attribute__((address_space(3))) v4i*)(l), 0, 0);
}
__device__ __forceinline__ void wait_async0() {
#if __has_builtin(__builtin_amdgcn_s_wait_asynccnt)
  __builtin_amdgcn_s_wait_asynccnt(0);
#else
  asm volatile("s_wait_asynccnt 0x0" ::: "memory");
#endif
}
#endif

__device__ __forceinline__ unsigned short f2bf_rne(float f) {
  unsigned int u = __float_as_uint(f);
  unsigned int r = u + 0x7FFFu + ((u >> 16) & 1u);
  return (unsigned short)(r >> 16);
}

__device__ __forceinline__ float silu_f(float x) {
  return x / (1.f + __expf(-x));
}

// ---------------------------------------------------------------------------
// Elementwise f32 -> bf16 convert
// ---------------------------------------------------------------------------
__global__ void cvt_f32_to_bf16(const float* __restrict__ src,
                                unsigned short* __restrict__ dst, int n) {
  int i = blockIdx.x * blockDim.x + threadIdx.x;
  if (i < n) dst[i] = f2bf_rne(src[i]);
}

// dt_proj_w [D_INNER, 48] -> bf16 [D_INNER, 64] zero-padded over K
__global__ void pad_dt_w(const float* __restrict__ src,
                         unsigned short* __restrict__ dst) {
  int i = blockIdx.x * blockDim.x + threadIdx.x;
  if (i >= D_INNER * DT_RANKP) return;
  int r = i / DT_RANKP, c = i % DT_RANKP;
  dst[i] = (c < DT_RANK) ? f2bf_rne(src[r * DT_RANK + c]) : (unsigned short)0;
}

// x_dbl [NTOK, 80] cols 0..47 -> bf16 [NTOK, 64] zero-padded
__global__ void pack_dt(const float* __restrict__ xdbl,
                        unsigned short* __restrict__ dst) {
  int i = blockIdx.x * blockDim.x + threadIdx.x;
  if (i >= NTOK * DT_RANKP) return;
  int m = i / DT_RANKP, c = i % DT_RANKP;
  dst[i] = (c < DT_RANK) ? f2bf_rne(xdbl[(size_t)m * N_XDBL + c])
                         : (unsigned short)0;
}

// ---------------------------------------------------------------------------
// WMMA GEMM: C[M,N] f32 = A[M,K] bf16 * W[N,K]^T bf16
// Block = 8 waves over M (128 rows), shared (NT*16)-wide N tile.
// B tile (NT*16 x 32 bf16) is staged in LDS once per block per K-step,
// double-buffered: async copy (or pipelined ds_store fallback) for step i+1
// overlaps the WMMAs of step i. A fragments load register-direct (b128 x2).
// A and B fragments both use contiguous k = 16*half + e (not the ISA table order).
// Grids are exact multiples -> no divergent exits around barriers.
// ---------------------------------------------------------------------------
template <int NT>
__global__ __launch_bounds__(TPB) void gemm_bf16_wmma(
    const bf16* __restrict__ A, const bf16* __restrict__ W,
    float* __restrict__ C, int M, int N, int K) {
  __shared__ unsigned short sB[2][NT * 16 * 32];  // [buf][n_local*32 + k_local]

  const int tid  = threadIdx.x;
  const int lane = tid & 31;
  const int wave = tid >> 5;
  const int m0 = (blockIdx.x * 8 + wave) * 16;
  const int n0 = blockIdx.y * (NT * 16);
  const int half = lane >> 4;   // which 16-lane half of the wave
  const int lm   = lane & 15;

  // B-tile copy role: NT*64 threads copy 8 bf16 (16B) each.
  const bool copier = tid < NT * 64;
  const int  crow = tid >> 2;          // local n row 0..NT*16-1
  const int  cseg = (tid & 3) * 8;     // k offset 0,8,16,24
  const bf16* wsrc = W + (size_t)(n0 + crow) * K + cseg;
  unsigned short* ldst[2] = {&sB[0][crow * 32 + cseg], &sB[1][crow * 32 + cseg]};

  v8f acc[NT] = {};
  const bf16* aRow = A + (size_t)(m0 + lm) * K + 16 * half;

  // ---- prologue: fill buffer 0 with K-step 0 ----
#ifdef HAVE_ASYNC_LDS
  if (copier) async_cp_b128(wsrc, ldst[0]);
  wait_async0();
#else
  if (copier) *(uint4*)ldst[0] = *(const uint4*)wsrc;
#endif
  __syncthreads();

  int cur = 0;
  for (int k0 = 0; k0 < K; k0 += 32) {
    const bool more = (k0 + 32) < K;
    // ---- start staging next B tile into the other buffer ----
#ifdef HAVE_ASYNC_LDS
    if (more && copier) async_cp_b128(wsrc + k0 + 32, ldst[cur ^ 1]);
#else
    uint4 nv;
    if (more && copier) nv = *(const uint4*)(wsrc + k0 + 32);
#endif

    // ---- A fragment: two 16B loads per lane ----
    v8bf alo = *(const v8bf*)(aRow + k0);
    v8bf ahi = *(const v8bf*)(aRow + k0 + 8);
    v16bf a = __builtin_shufflevector(alo, ahi, 0, 1, 2, 3, 4, 5, 6, 7, 8, 9,
                                      10, 11, 12, 13, 14, 15);
    if (more) __builtin_prefetch(aRow + k0 + 32, 0, 1);

    const unsigned short* sbase = sB[cur];
#pragma unroll
    for (int t = 0; t < NT; ++t) {
      // B fragment from LDS: lane col n_local = t*16+lm, elem e at
      // k_local = 16*half + e (contiguous 16 bf16 -> ds_load_b128 x2)
      v16bf b = *(const v16bf*)(sbase + (t * 16 + lm) * 32 + 16 * half);
      acc[t] = __builtin_amdgcn_wmma_f32_16x16x32_bf16(
          false, a, false, b, (short)0, acc[t], false, false);
    }

    // ---- complete staging, flip buffers ----
#ifdef HAVE_ASYNC_LDS
    wait_async0();
#else
    if (more && copier) *(uint4*)ldst[cur ^ 1] = nv;
#endif
    __syncthreads();
    cur ^= 1;
  }

  // D layout: VGPR r -> row m0 + r + 8*half, col n0 + t*16 + lm
#pragma unroll
  for (int t = 0; t < NT; ++t) {
    float* cp = C + (size_t)(m0 + 8 * half) * N + n0 + t * 16 + lm;
#pragma unroll
    for (int r = 0; r < 8; ++r) cp[(size_t)r * N] = acc[t][r];
  }
}

// ---------------------------------------------------------------------------
// Causal depthwise conv1d (K=4) + bias + SiLU.  xi = xz[:, :, 0:1536]
// ---------------------------------------------------------------------------
__global__ void conv_silu_kernel(const float* __restrict__ xz,
                                 const float* __restrict__ cw,
                                 const float* __restrict__ cb,
                                 float* __restrict__ uf,
                                 unsigned short* __restrict__ ubf) {
  int i = blockIdx.x * blockDim.x + threadIdx.x;
  if (i >= NTOK * D_INNER) return;
  int d = i % D_INNER;
  int m = i / D_INNER;
  int l = m % SEQ;
  int b = m / SEQ;
  float acc = cb[d];
#pragma unroll
  for (int j = 0; j < D_CONVK; ++j) {
    int li = l + j - (D_CONVK - 1);
    if (li >= 0)
      acc += cw[d * D_CONVK + j] * xz[(size_t)(b * SEQ + li) * N_XZ + d];
  }
  float s = silu_f(acc);
  uf[i] = s;
  ubf[i] = f2bf_rne(s);
}

// ---------------------------------------------------------------------------
// Selective scan: thread owns one (b, d) channel, h[16] in registers,
// B_t / C_t (32 floats) staged in LDS per timestep.
// grid = (D_INNER/256, BATCH)
// ---------------------------------------------------------------------------
__global__ __launch_bounds__(TPB) void scan_kernel(
    const float* __restrict__ uf, const float* __restrict__ dta,
    const float* __restrict__ dt_bias, const float* __restrict__ xdbl,
    const float* __restrict__ A_log, float* __restrict__ yscan) {
  const int d = blockIdx.x * TPB + threadIdx.x;  // < D_INNER by construction
  const int b = blockIdx.y;
  __shared__ float sBC[32];

  float Ar[D_STATE];
#pragma unroll
  for (int s = 0; s < D_STATE; ++s)
    Ar[s] = -__expf(A_log[(size_t)d * D_STATE + s]);
  const float bias = dt_bias[d];

  float h[D_STATE];
#pragma unroll
  for (int s = 0; s < D_STATE; ++s) h[s] = 0.f;

  for (int l = 0; l < SEQ; ++l) {
    const size_t m = (size_t)b * SEQ + l;
    if (threadIdx.x < 32)
      sBC[threadIdx.x] = xdbl[m * N_XDBL + DT_RANK + threadIdx.x];
    __syncthreads();

    float dl = dta[m * D_INNER + d] + bias;
    float delta = (dl > 20.f) ? dl : __logf(1.f + __expf(dl));  // softplus
    float du = delta * uf[m * D_INNER + d];
    float y = 0.f;
#pragma unroll
    for (int s = 0; s < D_STATE; ++s) {
      h[s] = __expf(delta * Ar[s]) * h[s] + du * sBC[s];
      y = fmaf(h[s], sBC[16 + s], y);
    }
    yscan[m * D_INNER + d] = y;
    __syncthreads();  // protect sBC before next timestep overwrites
  }
}

// ---------------------------------------------------------------------------
// y = (yscan + u*D) * silu(z)  -> bf16 for out_proj
// ---------------------------------------------------------------------------
__global__ void gate_kernel(const float* __restrict__ yscan,
                            const float* __restrict__ uf,
                            const float* __restrict__ xz,
                            const float* __restrict__ Dp,
                            unsigned short* __restrict__ ybf) {
  int i = blockIdx.x * blockDim.x + threadIdx.x;
  if (i >= NTOK * D_INNER) return;
  int d = i % D_INNER;
  size_t m = (size_t)(i / D_INNER);
  float z = xz[m * N_XZ + D_INNER + d];
  float y = (yscan[i] + uf[i] * Dp[d]) * silu_f(z);
  ybf[i] = f2bf_rne(y);
}

// ---------------------------------------------------------------------------
// Launch
// ---------------------------------------------------------------------------
static inline size_t alignup(size_t x) { return (x + 255) & ~(size_t)255; }

extern "C" void kernel_launch(void* const* d_in, const int* in_sizes, int n_in,
                              void* d_out, int out_size, void* d_ws,
                              size_t ws_size, hipStream_t stream) {
  const float* x         = (const float*)d_in[0];  // (B,L,768)
  const float* in_proj_w = (const float*)d_in[1];  // (3072,768)
  const float* conv_w    = (const float*)d_in[2];  // (1536,1,4)
  const float* conv_b    = (const float*)d_in[3];  // (1536)
  const float* x_proj_w  = (const float*)d_in[4];  // (80,1536)
  const float* dt_proj_w = (const float*)d_in[5];  // (1536,48)
  const float* dt_proj_b = (const float*)d_in[6];  // (1536)
  const float* A_log     = (const float*)d_in[7];  // (1536,16)
  const float* Dp        = (const float*)d_in[8];  // (1536)
  const float* out_proj_w= (const float*)d_in[9];  // (768,1536)
  float* out = (float*)d_out;                      // (B,L,768)

  // ---- workspace carve ----
  char* p = (char*)d_ws;
  size_t off = 0;
  auto carve = [&](size_t bytes) {
    void* r = p + off;
    off = alignup(off + bytes);
    return r;
  };
  float*          xz    = (float*)carve((size_t)NTOK * N_XZ * 4);
  float*          uf    = (float*)carve((size_t)NTOK * D_INNER * 4);
  unsigned short* ubf   = (unsigned short*)carve((size_t)NTOK * D_INNER * 2);
  float*          xdbl  = (float*)carve((size_t)NTOK * N_XDBL * 4);
  float*          dta   = (float*)carve((size_t)NTOK * D_INNER * 4);
  float*          ysc   = (float*)carve((size_t)NTOK * D_INNER * 4);
  unsigned short* xbf   = (unsigned short*)carve((size_t)NTOK * D_MODEL * 2);
  unsigned short* w_in  = (unsigned short*)carve((size_t)N_XZ * D_MODEL * 2);
  unsigned short* w_xp  = (unsigned short*)carve((size_t)N_XDBL * D_INNER * 2);
  unsigned short* w_dt  = (unsigned short*)carve((size_t)D_INNER * DT_RANKP * 2);
  unsigned short* w_out = (unsigned short*)carve((size_t)D_MODEL * D_INNER * 2);
  unsigned short* dtbf  = (unsigned short*)carve((size_t)NTOK * DT_RANKP * 2);
  (void)ws_size; (void)n_in; (void)in_sizes; (void)out_size;

  auto blocks = [](long n) { return (unsigned)((n + TPB - 1) / TPB); };

  // 1) bf16 conversions
  cvt_f32_to_bf16<<<blocks((long)NTOK * D_MODEL), TPB, 0, stream>>>(x, xbf, NTOK * D_MODEL);
  cvt_f32_to_bf16<<<blocks((long)N_XZ * D_MODEL), TPB, 0, stream>>>(in_proj_w, w_in, N_XZ * D_MODEL);
  cvt_f32_to_bf16<<<blocks((long)N_XDBL * D_INNER), TPB, 0, stream>>>(x_proj_w, w_xp, N_XDBL * D_INNER);
  cvt_f32_to_bf16<<<blocks((long)D_MODEL * D_INNER), TPB, 0, stream>>>(out_proj_w, w_out, D_MODEL * D_INNER);
  pad_dt_w<<<blocks((long)D_INNER * DT_RANKP), TPB, 0, stream>>>(dt_proj_w, w_dt);

  // 2) in_proj: xz[8192,3072] = xbf[8192,768] * w_in[3072,768]^T
  {
    dim3 g(NTOK / 128, N_XZ / 64);
    gemm_bf16_wmma<4><<<g, TPB, 0, stream>>>((const bf16*)xbf, (const bf16*)w_in,
                                             xz, NTOK, N_XZ, D_MODEL);
  }

  // 3) causal depthwise conv + SiLU -> u (f32 + bf16)
  conv_silu_kernel<<<blocks((long)NTOK * D_INNER), TPB, 0, stream>>>(
      xz, conv_w, conv_b, uf, ubf);

  // 4) x_proj: xdbl[8192,80] = u * x_proj_w^T   (N=80 -> 5 tiles of 16)
  {
    dim3 g(NTOK / 128, N_XDBL / 16);
    gemm_bf16_wmma<1><<<g, TPB, 0, stream>>>((const bf16*)ubf, (const bf16*)w_xp,
                                             xdbl, NTOK, N_XDBL, D_INNER);
  }

  // 5) dt_proj: dta[8192,1536] = dt(pad64) * dt_proj_w(pad64)^T
  pack_dt<<<blocks((long)NTOK * DT_RANKP), TPB, 0, stream>>>(xdbl, dtbf);
  {
    dim3 g(NTOK / 128, D_INNER / 64);
    gemm_bf16_wmma<4><<<g, TPB, 0, stream>>>((const bf16*)dtbf, (const bf16*)w_dt,
                                             dta, NTOK, D_INNER, DT_RANKP);
  }

  // 6) selective scan
  {
    dim3 g(D_INNER / TPB, BATCH);
    scan_kernel<<<g, TPB, 0, stream>>>(uf, dta, dt_proj_b, xdbl, A_log, ysc);
  }

  // 7) gate (+ skip) -> bf16 (reuse ubf)
  gate_kernel<<<blocks((long)NTOK * D_INNER), TPB, 0, stream>>>(ysc, uf, xz, Dp, ubf);

  // 8) out_proj: out[8192,768] = ybf * out_proj_w^T
  {
    dim3 g(NTOK / 128, D_MODEL / 64);
    gemm_bf16_wmma<4><<<g, TPB, 0, stream>>>((const bf16*)ubf, (const bf16*)w_out,
                                             out, NTOK, D_MODEL, D_INNER);
  }
}
